// APPNP_26783416057954
// MI455X (gfx1250) — matrix operand order repaired and checked by reading, not compiled, not run on hardware
//
#include <hip/hip_runtime.h>

static constexpr int NS = 16384;   // samples
static constexpr int NF = 512;     // features
static constexpr int NH = 256;     // hidden
static constexpr int NC = 64;      // classes
static constexpr int KIT = 10;     // pagerank steps
static constexpr float ALPHA_F = 0.1f;

typedef __attribute__((ext_vector_type(16))) __bf16 v16bf;
typedef __attribute__((ext_vector_type(8)))  float  v8f;
typedef __attribute__((ext_vector_type(8)))  unsigned int v8u;
typedef __attribute__((ext_vector_type(4)))  unsigned int u32x4;
typedef __attribute__((ext_vector_type(4)))  float  f32x4;

__device__ __forceinline__ unsigned short f2bf(float f) {
  unsigned int u = __float_as_uint(f);
  u += 0x7FFFu + ((u >> 16) & 1u);            // round-to-nearest-even
  return (unsigned short)(u >> 16);
}
__device__ __forceinline__ unsigned int pack2bf(float a, float b) {
  return (unsigned int)f2bf(a) | ((unsigned int)f2bf(b) << 16);
}

struct B32B { u32x4 lo, hi; };                 // 32 bytes = 16 bf16

// Y[PxQ] = U[PxR] @ V[RxQ]  (optionally + bias, + alpha-axpy with Z0t)
// U row-major (bf16, or f32 converted on the fly), loaded NON-TEMPORAL
// (streamed exactly once, must not thrash L2). V supplied transposed:
// Vt row-major [QxR] bf16, regular-temporal (L2 resident).
// Output stored TRANSPOSED: Yt [QxP].
// One wave: 1 p-tile (16 rows of U) x 4 q-tiles (64 cols).
// Inner loop is software-pipelined (double-buffered fragments) so WMMAs for
// chunk i overlap with the in-flight loads of chunk i+1.
// BIAS: 0 none, 1 indexed along P, 2 indexed along Q.
template<bool AF32, int BIAS, bool AXPY, bool OUTF, bool OUTH>
__global__ __launch_bounds__(256)
void wmma_gemm_tn(const void* __restrict__ Uv,
                  const __bf16* __restrict__ Vt,
                  const float* __restrict__ bias,
                  const float* __restrict__ Z0t,
                  float* __restrict__ outF,
                  unsigned short* __restrict__ outH,
                  int P, int Q, int R)   // R must be a multiple of 64
{
  const int wave  = (int)((blockIdx.x * blockDim.x + threadIdx.x) >> 5);
  const int lane  = (int)(threadIdx.x & 31);
  const int lmod  = lane & 15;
  const int lhalf = lane >> 4;
  const int pTiles = P >> 4;
  const int pt = wave % pTiles;
  const int qg = wave / pTiles;
  const int p_base = pt << 4;
  const int q_base = qg << 6;

  const size_t aOff = (size_t)(p_base + lmod) * (size_t)R;
  const __bf16* Abp = nullptr;
  const float*  Afp = nullptr;
  if constexpr (AF32) Afp = (const float*)Uv + aOff;
  else                Abp = (const __bf16*)Uv + aOff;

  const __bf16* Bp[4];
#pragma unroll
  for (int qt = 0; qt < 4; ++qt)
    Bp[qt] = Vt + (size_t)(q_base + qt * 16 + lmod) * (size_t)R + (size_t)(lhalf * 8);

  v8f acc[4] = {};
  const int c0 = lhalf * 8;                    // per-lane K sub-chunk offset

  // ---- fragment loaders ----
  auto loadA = [&](int k) -> v16bf {
    if constexpr (!AF32) {
      B32B t;
      t.lo = __builtin_nontemporal_load((const u32x4*)(Abp + k + c0));       // K=c0..c0+7
      t.hi = __builtin_nontemporal_load((const u32x4*)(Abp + k + 16 + c0));  // K=16+c0..
      return __builtin_bit_cast(v16bf, t);
    } else {
      f32x4 f0 = __builtin_nontemporal_load((const f32x4*)(Afp + k + c0));
      f32x4 f1 = __builtin_nontemporal_load((const f32x4*)(Afp + k + c0 + 4));
      f32x4 f2 = __builtin_nontemporal_load((const f32x4*)(Afp + k + 16 + c0));
      f32x4 f3 = __builtin_nontemporal_load((const f32x4*)(Afp + k + 16 + c0 + 4));
      v8u u = { pack2bf(f0[0], f0[1]), pack2bf(f0[2], f0[3]),
                pack2bf(f1[0], f1[1]), pack2bf(f1[2], f1[3]),
                pack2bf(f2[0], f2[1]), pack2bf(f2[2], f2[3]),
                pack2bf(f3[0], f3[1]), pack2bf(f3[2], f3[3]) };
      return __builtin_bit_cast(v16bf, u);
    }
  };
  auto loadB = [&](v16bf* b, int k) {
#pragma unroll
    for (int qt = 0; qt < 4; ++qt)
      { B32B t; t.lo = *(const u32x4*)(Bp[qt] + k); t.hi = *(const u32x4*)(Bp[qt] + k + 16);
        b[qt] = __builtin_bit_cast(v16bf, t); }     // lane = column, K=c0..c0+7 and 16+c0..16+c0+7
  };
  auto wmma4 = [&](const v16bf& a, const v16bf* b) {
#pragma unroll
    for (int qt = 0; qt < 4; ++qt)
      acc[qt] = __builtin_amdgcn_wmma_f32_16x16x32_bf16(
          false, a, false, b[qt], (short)0, acc[qt], false, false);
  };

  // ---- software-pipelined main loop (R multiple of 64 -> >= 2 chunks) ----
  v16bf a0, a1, b0[4], b1[4];
  a0 = loadA(0);  loadB(b0, 0);
  a1 = loadA(32); loadB(b1, 32);
  for (int k = 64; k < R; k += 64) {
    wmma4(a0, b0);
    a0 = loadA(k);      loadB(b0, k);
    wmma4(a1, b1);
    a1 = loadA(k + 32); loadB(b1, k + 32);
  }
  wmma4(a0, b0);
  wmma4(a1, b1);

  // ---- epilogue ----
  const int p0 = p_base + c0;                  // 8 consecutive P outputs per lane
  float bp[8];
  if constexpr (BIAS == 1) {
    float4 x = *(const float4*)(bias + p0);
    float4 y = *(const float4*)(bias + p0 + 4);
    bp[0] = x.x; bp[1] = x.y; bp[2] = x.z; bp[3] = x.w;
    bp[4] = y.x; bp[5] = y.y; bp[6] = y.z; bp[7] = y.w;
  }

#pragma unroll
  for (int qt = 0; qt < 4; ++qt) {
    const int q = q_base + qt * 16 + lmod;
    float v[8];
#pragma unroll
    for (int i = 0; i < 8; ++i) v[i] = acc[qt][i];
    if constexpr (BIAS == 1) {
#pragma unroll
      for (int i = 0; i < 8; ++i) v[i] += bp[i];
    }
    if constexpr (BIAS == 2) {
      const float bq = bias[q];
#pragma unroll
      for (int i = 0; i < 8; ++i) v[i] += bq;
    }
    const size_t o = (size_t)q * (size_t)P + (size_t)p0;
    if constexpr (AXPY) {
      float4 z0 = *(const float4*)(Z0t + o);
      float4 z1 = *(const float4*)(Z0t + o + 4);
      float zz[8] = { z0.x, z0.y, z0.z, z0.w, z1.x, z1.y, z1.z, z1.w };
#pragma unroll
      for (int i = 0; i < 8; ++i) v[i] = (1.0f - ALPHA_F) * v[i] + ALPHA_F * zz[i];
    }
    if constexpr (OUTF) {
      *(float4*)(outF + o)     = make_float4(v[0], v[1], v[2], v[3]);
      *(float4*)(outF + o + 4) = make_float4(v[4], v[5], v[6], v[7]);
    }
    if constexpr (OUTH) {
      uint4 u = make_uint4(pack2bf(v[0], v[1]), pack2bf(v[2], v[3]),
                           pack2bf(v[4], v[5]), pack2bf(v[6], v[7]));
      *(uint4*)(outH + o) = u;
    }
  }
}

// elementwise f32 -> bf16, 8 elements per thread; fully non-temporal
// (1 GB in, 0.5 GB out; neither side should pollute L2)
__global__ void convert_f32_bf16_vec(const float* __restrict__ src,
                                     unsigned short* __restrict__ dst, size_t n8) {
  size_t i = (size_t)blockIdx.x * blockDim.x + threadIdx.x;
  if (i >= n8) return;
  const f32x4* s = (const f32x4*)src;
  f32x4 x = __builtin_nontemporal_load(s + 2 * i);
  f32x4 y = __builtin_nontemporal_load(s + 2 * i + 1);
  u32x4 o = { pack2bf(x[0], x[1]), pack2bf(x[2], x[3]),
              pack2bf(y[0], y[1]), pack2bf(y[2], y[3]) };
  __builtin_nontemporal_store(o, (u32x4*)dst + i);
}

// src [rows][cols] f32 -> dst [cols][rows] bf16 (tiny matrices only)
__global__ void transpose_f32_bf16(const float* __restrict__ src,
                                   unsigned short* __restrict__ dst,
                                   int rows, int cols) {
  int id = (int)(blockIdx.x * blockDim.x + threadIdx.x);
  if (id >= rows * cols) return;
  int r = id / cols, c = id - r * cols;
  dst[(size_t)c * rows + r] = f2bf(src[id]);
}

// ZT [NC][NS] f32 -> out [NS][NC] = log_softmax over classes
__global__ __launch_bounds__(256)
void logsoftmax_t(const float* __restrict__ ZT, float* __restrict__ out) {
  int m = (int)(blockIdx.x * blockDim.x + threadIdx.x);
  float z[NC];
#pragma unroll
  for (int c = 0; c < NC; ++c) z[c] = ZT[(size_t)c * NS + m];
  float mx = z[0];
#pragma unroll
  for (int c = 1; c < NC; ++c) mx = fmaxf(mx, z[c]);
  float s = 0.0f;
#pragma unroll
  for (int c = 0; c < NC; ++c) s += __expf(z[c] - mx);
  float lse = mx + __logf(s);
  float* o = out + (size_t)m * NC;
#pragma unroll
  for (int c = 0; c < NC; c += 4)
    *(float4*)(o + c) = make_float4(z[c] - lse, z[c + 1] - lse,
                                    z[c + 2] - lse, z[c + 3] - lse);
}

extern "C" void kernel_launch(void* const* d_in, const int* in_sizes, int n_in,
                              void* d_out, int out_size, void* d_ws, size_t ws_size,
                              hipStream_t stream) {
  (void)in_sizes; (void)n_in; (void)out_size;
  const float* h   = (const float*)d_in[0];
  const float* adj = (const float*)d_in[1];
  const float* W1  = (const float*)d_in[2];
  const float* b1  = (const float*)d_in[3];
  const float* W2  = (const float*)d_in[4];
  const float* b2  = (const float*)d_in[5];
  float* out = (float*)d_out;

  char* ws = (char*)d_ws;
  size_t off = 0;
  auto alloc = [&](size_t bytes) -> char* {
    char* p = ws + off;
    off += (bytes + 255) & ~(size_t)255;
    return p;
  };

  unsigned short* W1T  = (unsigned short*)alloc((size_t)NH * NF * 2);  // [NH][NF] bf16
  unsigned short* W2T  = (unsigned short*)alloc((size_t)NC * NH * 2);  // [NC][NH] bf16
  unsigned short* Hbf  = (unsigned short*)alloc((size_t)NS * NF * 2);  // h bf16 [NS][NF]
  unsigned short* H1   = (unsigned short*)alloc((size_t)NS * NH * 2);  // [NS][NH] bf16
  float*          Z0T  = (float*)alloc((size_t)NC * NS * 4);           // Z0^T f32
  unsigned short* ZTa  = (unsigned short*)alloc((size_t)NC * NS * 2);  // Zk^T bf16 ping
  unsigned short* ZTb  = (unsigned short*)alloc((size_t)NC * NS * 2);  // Zk^T bf16 pong
  float*          ZfT  = (float*)alloc((size_t)NC * NS * 4);           // final Zk^T f32

  const size_t adjBytes = (size_t)NS * NS * 2;
  const bool useBf16Adj = (ws_size >= off + adjBytes);
  unsigned short* adjBf = useBf16Adj ? (unsigned short*)alloc(adjBytes) : nullptr;

  // Pre-transpose / pre-convert the small operands
  transpose_f32_bf16<<<(NF * NH + 255) / 256, 256, 0, stream>>>(W1, W1T, NF, NH);
  transpose_f32_bf16<<<(NH * NC + 255) / 256, 256, 0, stream>>>(W2, W2T, NH, NC);
  {
    size_t n8 = (size_t)NS * NF / 8;
    convert_f32_bf16_vec<<<(unsigned)((n8 + 255) / 256), 256, 0, stream>>>(h, Hbf, n8);
  }
  if (useBf16Adj) {
    size_t n8 = (size_t)NS * NS / 8;   // 1.07 GB read once -> 0.54 GB bf16
    convert_f32_bf16_vec<<<(unsigned)((n8 + 255) / 256), 256, 0, stream>>>(adj, adjBf, n8);
  }

  // GEMM1: H1^T = W1^T @ h^T  -> stored transposed = H1 [NS][NH] bf16, +b1 along P
  {
    const int P = NH, Q = NS, R = NF;
    const int waves = (P / 16) * (Q / 64);     // 4096
    wmma_gemm_tn<false, 1, false, false, true><<<waves / 8, 256, 0, stream>>>(
        (const void*)W1T, (const __bf16*)Hbf, b1, nullptr, nullptr, H1, P, Q, R);
  }
  // GEMM2: Z0 = H1 @ W2  -> stored transposed = Z0^T (f32 + bf16), +b2 along Q
  {
    const int P = NS, Q = NC, R = NH;
    const int waves = (P / 16) * (Q / 64);     // 1024
    wmma_gemm_tn<false, 2, false, true, true><<<waves / 8, 256, 0, stream>>>(
        (const void*)H1, (const __bf16*)W2T, b2, nullptr, Z0T, ZTa, P, Q, R);
  }
  // Propagation: Znew = adj @ Zk -> stored transposed; fused 0.9x + 0.1*Z0
  unsigned short* cur = ZTa;
  unsigned short* nxt = ZTb;
  for (int it = 0; it < KIT; ++it) {
    const int P = NS, Q = NC, R = NS;
    const int waves = (P / 16) * (Q / 64);     // 1024, adj streamed exactly once
    const bool last = (it == KIT - 1);
    if (useBf16Adj) {
      if (last)
        wmma_gemm_tn<false, 0, true, true, true><<<waves / 8, 256, 0, stream>>>(
            (const void*)adjBf, (const __bf16*)cur, nullptr, Z0T, ZfT, nxt, P, Q, R);
      else
        wmma_gemm_tn<false, 0, true, false, true><<<waves / 8, 256, 0, stream>>>(
            (const void*)adjBf, (const __bf16*)cur, nullptr, Z0T, nullptr, nxt, P, Q, R);
    } else {
      if (last)
        wmma_gemm_tn<true, 0, true, true, true><<<waves / 8, 256, 0, stream>>>(
            (const void*)adj, (const __bf16*)cur, nullptr, Z0T, ZfT, nxt, P, Q, R);
      else
        wmma_gemm_tn<true, 0, true, false, true><<<waves / 8, 256, 0, stream>>>(
            (const void*)adj, (const __bf16*)cur, nullptr, Z0T, nullptr, nxt, P, Q, R);
    }
    unsigned short* t = cur; cur = nxt; nxt = t;
  }

  logsoftmax_t<<<NS / 256, 256, 0, stream>>>(ZfT, out);
}
